// MultiScaleHeteroConv_38397007626774
// MI455X (gfx1250) — matrix operand order repaired and checked by reading, not compiled, not run on hardware
//
#include <hip/hip_runtime.h>
#include <hip/hip_bf16.h>

typedef __attribute__((ext_vector_type(16))) __bf16 bf16x16;
typedef __attribute__((ext_vector_type(8)))  __bf16 bf16x8;
typedef __attribute__((ext_vector_type(8)))  float  f32x8;

#define F    128
#define HEADS 4
#define CH    32

// ---------------------------------------------------------------------------
// float atomic max via signed/unsigned integer ordering trick (valid with
// -inf initialization; correct for mixed-sign candidates).
// ---------------------------------------------------------------------------
__device__ __forceinline__ void atomicMaxF(float* addr, float val) {
  if (val >= 0.0f)
    atomicMax((int*)addr, __float_as_int(val));
  else
    atomicMin((unsigned int*)addr, (unsigned int)__float_as_int(val));
}

__device__ __forceinline__ void relw2(const float* rg, float& w0, float& w1) {
  if (rg == nullptr) { w0 = 1.0f; w1 = 0.0f; return; }
  float g0 = rg[0], g1 = rg[1];
  float mx = fmaxf(g0, g1);
  float e0 = __expf(g0 - mx), e1 = __expf(g1 - mx);
  float inv = 1.0f / (e0 + e1);
  w0 = e0 * inv; w1 = e1 * inv;
}

// ---------------------------------------------------------------------------
// bf16 WMMA GEMM: C[M,128] = A[M,K](bf16,row-major) @ B[K,128], where B is
// supplied pre-transposed as Bt[n][k] (bf16). One wave per 16x16 output tile.
// K must be a multiple of 32 (128 or 256 here). Optional bias[128] epilogue.
// VGPR layouts: 16-bit A/B both in contiguous k order (k = 16*half + s); f32 C/D per CDNA5 ISA 7.12.2.
// ---------------------------------------------------------------------------
__global__ __launch_bounds__(256) void gemm_bf16(
    const __bf16* __restrict__ A, const __bf16* __restrict__ Bt,
    const float* __restrict__ bias, float* __restrict__ Cm, int M, int K)
{
  const int wave    = (int)((blockIdx.x * blockDim.x + threadIdx.x) >> 5);
  const int lane    = (int)(threadIdx.x & 31);
  const int rowTile = wave >> 3;      // 8 col tiles of 16 cover 128 outputs
  const int colTile = wave & 7;
  const int r0      = rowTile * 16;
  if (r0 >= M) return;                 // uniform per wave -> EXEC all-ones
  const int half = lane >> 4;          // lane group 0: lanes 0-15, 1: 16-31
  const int lid  = lane & 15;

  int arow = r0 + lid; if (arow > M - 1) arow = M - 1;   // tail clamp (stores guarded)
  const __bf16* Arow = A + (size_t)arow * K;
  const int ncol = colTile * 16 + lid;
  const __bf16* Bcol = Bt + (size_t)ncol * K;

  f32x8 acc = {};
  const int kchunks = K >> 5;
  for (int kc = 0; kc < kchunks; ++kc) {
    const int kb = kc << 5;
    // A 16x32 bf16: lanes 0-15 hold K kb+0..15; lanes 16-31 hold
    // K kb+16..31; packed 2 per VGPR.
    bf16x8 alo = *(const bf16x8*)(Arow + kb + half * 16);
    bf16x8 ahi = *(const bf16x8*)(Arow + kb + half * 16 + 8);
    bf16x16 a;
#pragma unroll
    for (int i = 0; i < 8; ++i) { a[i] = alo[i]; a[8 + i] = ahi[i]; }
    // B 32x16 bf16: lane = column n; lanes 0-15 hold K kb+0..15, lanes 16-31
    // hold K kb+16..31 (contiguous in Bt[n][k]).
    bf16x16 b = *(const bf16x16*)(Bcol + kb + half * 16);
    acc = __builtin_amdgcn_wmma_f32_16x16x32_bf16(false, a, false, b,
                                                  (short)0, acc, false, false);
  }
  const float bv = bias ? bias[ncol] : 0.0f;
#pragma unroll
  for (int r = 0; r < 8; ++r) {
    int m = r0 + half * 8 + r;         // C/D: lanes 0-15 -> M 0..7, 16-31 -> 8..15
    if (m < M) Cm[(size_t)m * F + ncol] = acc[r] + bv;
  }
}

// ---------------------------------------------------------------------------
// Elementwise / conversion helpers
// ---------------------------------------------------------------------------
__global__ void f32_to_bf16_k(const float* __restrict__ x, __bf16* __restrict__ y, int n) {
  int t = blockIdx.x * blockDim.x + threadIdx.x;
  if (t < n) y[t] = (__bf16)x[t];
}

// W[K,128] row-major -> Wt[128,K] bf16
__global__ void w_transpose_bf16(const float* __restrict__ W, __bf16* __restrict__ Wt, int K) {
  int t = blockIdx.x * blockDim.x + threadIdx.x;
  if (t >= K * F) return;
  int k = t / F, n = t % F;
  Wt[(size_t)n * K + k] = (__bf16)W[t];
}

__global__ void fill_f32(float* __restrict__ p, float v, int n) {
  int t = blockIdx.x * blockDim.x + threadIdx.x;
  if (t < n) p[t] = v;
}

// a[n,h] = sum_c h[n, h*32+c] * att[h*32+c]
__global__ void att_dot(const float* __restrict__ h, const float* __restrict__ att,
                        float* __restrict__ a, int n) {
  int t = blockIdx.x * blockDim.x + threadIdx.x;
  if (t >= n * HEADS) return;
  int node = t >> 2, hd = t & 3;
  const float* hp = h + (size_t)node * F + hd * CH;
  const float* ap = att + hd * CH;
  float s = 0.0f;
#pragma unroll
  for (int c = 0; c < CH; ++c) s += hp[c] * ap[c];
  a[t] = s;
}

// ---------------------------------------------------------------------------
// Scatter-softmax over edges (3 phases) + weighted feature aggregation
// ---------------------------------------------------------------------------
__global__ void edge_max(const int* __restrict__ src, const int* __restrict__ dst,
                         int E0, int nSelf, const float* __restrict__ a_s,
                         const float* __restrict__ a_d, float slope,
                         float* __restrict__ e_out, float* __restrict__ m) {
  int e = blockIdx.x * blockDim.x + threadIdx.x;
  int Et = E0 + nSelf;
  if (e >= Et) return;
  int s = (e < E0) ? src[e] : (e - E0);
  int d = (e < E0) ? dst[e] : (e - E0);
#pragma unroll
  for (int h = 0; h < HEADS; ++h) {
    float v = a_s[s * HEADS + h] + a_d[d * HEADS + h];
    v = (v > 0.0f) ? v : v * slope;      // leaky relu
    e_out[(size_t)e * HEADS + h] = v;
    atomicMaxF(&m[d * HEADS + h], v);
  }
}

__global__ void fix_m(float* __restrict__ m, int n) {
  int t = blockIdx.x * blockDim.x + threadIdx.x;
  if (t < n) { float v = m[t]; if (!(v > -1e37f)) m[t] = 0.0f; }
}

__global__ void edge_expsum(const int* __restrict__ dst, int E0, int nSelf,
                            const float* __restrict__ m, float* __restrict__ w_e,
                            float* __restrict__ den) {
  int e = blockIdx.x * blockDim.x + threadIdx.x;
  int Et = E0 + nSelf;
  if (e >= Et) return;
  int d = (e < E0) ? dst[e] : (e - E0);
#pragma unroll
  for (int h = 0; h < HEADS; ++h) {
    float w = __expf(w_e[(size_t)e * HEADS + h] - m[d * HEADS + h]);
    w_e[(size_t)e * HEADS + h] = w;
    atomicAdd(&den[d * HEADS + h], w);
  }
}

// one wave32 per edge; lane = channel within head; relation-gate weight folded in
__global__ void edge_agg(const int* __restrict__ src, const int* __restrict__ dst,
                         int E0, int nSelf, const float* __restrict__ w_e,
                         const float* __restrict__ den, const float* __restrict__ hs,
                         const float* __restrict__ rel_gates, int gateIdx,
                         float* __restrict__ acc) {
  int gt = blockIdx.x * blockDim.x + threadIdx.x;
  int e = gt >> 5, lane = gt & 31;
  int Et = E0 + nSelf;
  if (e >= Et) return;
  int s = (e < E0) ? src[e] : (e - E0);
  int d = (e < E0) ? dst[e] : (e - E0);
  float w0, w1; relw2(rel_gates, w0, w1);
  float relw = (gateIdx <= 0) ? w0 : w1;      // gateIdx<0 -> w0==1
  const float* hrow = hs + (size_t)s * F;
  float* arow = acc + (size_t)d * F;
#pragma unroll
  for (int h = 0; h < HEADS; ++h) {
    float c = relw * w_e[(size_t)e * HEADS + h] / (den[d * HEADS + h] + 1e-16f);
    int f = h * CH + lane;
    atomicAdd(&arow[f], c * hrow[f]);
  }
}

// ---------------------------------------------------------------------------
// Fusion epilogue: cat = [loc, glb] (bf16); fused = sigmoid(pre)*loc + ...
// loc = accL + (w0*bL0 + w1*bL1), glb likewise (relation gates + gat biases).
// ---------------------------------------------------------------------------
__global__ void build_cat(const float* __restrict__ accL, const float* __restrict__ accG,
                          const float* __restrict__ bL0, const float* __restrict__ bL1,
                          const float* __restrict__ bG0, const float* __restrict__ bG1,
                          const float* __restrict__ rg, __bf16* __restrict__ cat, int n) {
  int t = blockIdx.x * blockDim.x + threadIdx.x;
  if (t >= n * F) return;
  int node = t >> 7, f = t & (F - 1);
  float w0, w1; relw2(rg, w0, w1);
  float bl = w0 * bL0[f] + (bL1 ? w1 * bL1[f] : 0.0f);
  float bg = w0 * bG0[f] + (bG1 ? w1 * bG1[f] : 0.0f);
  size_t co = (size_t)node * (2 * F) + f;
  cat[co]     = (__bf16)(accL[t] + bl);
  cat[co + F] = (__bf16)(accG[t] + bg);
}

__global__ void fuse_gate(const float* __restrict__ accL, const float* __restrict__ accG,
                          const float* __restrict__ bL0, const float* __restrict__ bL1,
                          const float* __restrict__ bG0, const float* __restrict__ bG1,
                          const float* __restrict__ rg, const float* __restrict__ gatePre,
                          __bf16* __restrict__ fused, int n) {
  int t = blockIdx.x * blockDim.x + threadIdx.x;
  if (t >= n * F) return;
  int f = t & (F - 1);
  float w0, w1; relw2(rg, w0, w1);
  float loc = accL[t] + w0 * bL0[f] + (bL1 ? w1 * bL1[f] : 0.0f);
  float glb = accG[t] + w0 * bG0[f] + (bG1 ? w1 * bG1[f] : 0.0f);
  float g = 1.0f / (1.0f + __expf(-gatePre[t]));
  fused[t] = (__bf16)(g * loc + (1.0f - g) * glb);
}

// ---------------------------------------------------------------------------
extern "C" void kernel_launch(void* const* d_in, const int* in_sizes, int n_in,
                              void* d_out, int out_size, void* d_ws, size_t ws_size,
                              hipStream_t stream) {
  const int N = in_sizes[0] / F;
  const int E = in_sizes[16] / 2;

  const float* x_user    = (const float*)d_in[0];
  const float* x_item    = (const float*)d_in[1];
  const float* W_src_l   = (const float*)d_in[2];
  const float* W_dst_l   = (const float*)d_in[3];
  const float* att_src_l = (const float*)d_in[4];
  const float* att_dst_l = (const float*)d_in[5];
  const float* bias_l    = (const float*)d_in[6];
  const float* W_src_g   = (const float*)d_in[7];
  const float* W_dst_g   = (const float*)d_in[8];
  const float* att_src_g = (const float*)d_in[9];
  const float* att_dst_g = (const float*)d_in[10];
  const float* bias_g    = (const float*)d_in[11];
  const float* gate_W    = (const float*)d_in[12];
  const float* gate_b    = (const float*)d_in[13];
  const float* out_W     = (const float*)d_in[14];
  const float* rel_gates = (const float*)d_in[15];
  const int*   ei_buys   = (const int*)d_in[16];
  const int*   ei_rev    = (const int*)d_in[17];
  const int*   ei_fol    = (const int*)d_in[18];

  // ---- workspace carve-out ----
  char* p = (char*)d_ws;
  auto alloc = [&](size_t bytes) -> char* {
    char* r = p; p += (bytes + 255) & ~(size_t)255; return r;
  };
  const size_t NF4 = (size_t)N * F * 4;
  const size_t NF2 = (size_t)N * F * 2;
  float* acc_ul = (float*)alloc(NF4);           // user local (rev+follows, gw-weighted)
  float* acc_ug = (float*)alloc(NF4);
  float* acc_il = (float*)alloc(NF4);           // item local (buys)
  float* acc_ig = (float*)alloc(NF4);
  float* h_s    = (float*)alloc(NF4);
  float* h_d    = (float*)alloc(NF4);
  float* gpre   = (float*)alloc(NF4);
  __bf16* xu_bf   = (__bf16*)alloc(NF2);
  __bf16* xi_bf   = (__bf16*)alloc(NF2);
  __bf16* cat_bf  = (__bf16*)alloc((size_t)N * 2 * F * 2);
  __bf16* fused_bf= (__bf16*)alloc(NF2);
  __bf16* Wt      = (__bf16*)alloc((size_t)12 * F * F * 2);
  __bf16* gWt     = (__bf16*)alloc((size_t)2 * 2 * F * F * 2);
  __bf16* oWt     = (__bf16*)alloc((size_t)2 * F * F * 2);
  float* a_s   = (float*)alloc((size_t)N * HEADS * 4);
  float* a_d   = (float*)alloc((size_t)N * HEADS * 4);
  float* m_buf = (float*)alloc((size_t)N * HEADS * 4);
  float* den   = (float*)alloc((size_t)N * HEADS * 4);
  float* w_e   = (float*)alloc((size_t)(E + N) * HEADS * 4);

  auto cdiv = [](long long a, long long b) { return (int)((a + b - 1) / b); };
  auto gemm = [&](const __bf16* A, const __bf16* Bt, const float* bias,
                  float* C, int M, int K) {
    long long waves = (long long)((M + 15) / 16) * 8;
    gemm_bf16<<<cdiv(waves * 32, 256), 256, 0, stream>>>(A, Bt, bias, C, M, K);
  };

  const int nx = N * F;

  // ---- one-time conversions ----
  f32_to_bf16_k<<<cdiv(nx, 256), 256, 0, stream>>>(x_user, xu_bf, nx);
  f32_to_bf16_k<<<cdiv(nx, 256), 256, 0, stream>>>(x_item, xi_bf, nx);
  for (int sc = 0; sc < 2; ++sc)
    for (int r = 0; r < 3; ++r) {
      const float* Ws = (sc == 0 ? W_src_l : W_src_g) + (size_t)r * F * F;
      const float* Wd = (sc == 0 ? W_dst_l : W_dst_g) + (size_t)r * F * F;
      w_transpose_bf16<<<cdiv(F * F, 256), 256, 0, stream>>>(
          Ws, Wt + (size_t)(sc * 6 + r * 2 + 0) * F * F, F);
      w_transpose_bf16<<<cdiv(F * F, 256), 256, 0, stream>>>(
          Wd, Wt + (size_t)(sc * 6 + r * 2 + 1) * F * F, F);
    }
  for (int t = 0; t < 2; ++t) {
    w_transpose_bf16<<<cdiv(2 * F * F, 256), 256, 0, stream>>>(
        gate_W + (size_t)t * 2 * F * F, gWt + (size_t)t * 2 * F * F, 2 * F);
    w_transpose_bf16<<<cdiv(F * F, 256), 256, 0, stream>>>(
        out_W + (size_t)t * F * F, oWt + (size_t)t * F * F, F);
  }
  fill_f32<<<cdiv(nx, 256), 256, 0, stream>>>(acc_ul, 0.0f, nx);
  fill_f32<<<cdiv(nx, 256), 256, 0, stream>>>(acc_ug, 0.0f, nx);
  fill_f32<<<cdiv(nx, 256), 256, 0, stream>>>(acc_il, 0.0f, nx);
  fill_f32<<<cdiv(nx, 256), 256, 0, stream>>>(acc_ig, 0.0f, nx);

  // ---- per (scale, relation) GAT pass ----
  struct Rel { const int* src; const int* dst; int E0; int nSelf;
               const __bf16* xs; const __bf16* xd; int dstUser; int gateIdx; };
  const Rel rels[3] = {
    { ei_buys, ei_buys + E, E, 0, xu_bf, xi_bf, 0, -1 },  // user -buys-> item
    { ei_rev,  ei_rev  + E, E, 0, xi_bf, xu_bf, 1,  0 },  // item -rev-> user
    { ei_fol,  ei_fol  + E, E, N, xu_bf, xu_bf, 1,  1 },  // user -follows-> user (+self loops)
  };
  const int nm = N * HEADS;
  for (int sc = 0; sc < 2; ++sc) {
    const float slope = (sc == 0) ? 0.2f : 0.3f;
    for (int r = 0; r < 3; ++r) {
      const Rel& R = rels[r];
      const __bf16* WtS = Wt + (size_t)(sc * 6 + r * 2 + 0) * F * F;
      const __bf16* WtD = Wt + (size_t)(sc * 6 + r * 2 + 1) * F * F;
      gemm(R.xs, WtS, nullptr, h_s, N, F);
      gemm(R.xd, WtD, nullptr, h_d, N, F);
      const float* attS = (sc == 0 ? att_src_l : att_src_g) + (size_t)r * HEADS * CH;
      const float* attD = (sc == 0 ? att_dst_l : att_dst_g) + (size_t)r * HEADS * CH;
      att_dot<<<cdiv(nm, 256), 256, 0, stream>>>(h_s, attS, a_s, N);
      att_dot<<<cdiv(nm, 256), 256, 0, stream>>>(h_d, attD, a_d, N);
      fill_f32<<<cdiv(nm, 256), 256, 0, stream>>>(m_buf, -__builtin_inff(), nm);
      fill_f32<<<cdiv(nm, 256), 256, 0, stream>>>(den, 0.0f, nm);
      const int Et = R.E0 + R.nSelf;
      edge_max<<<cdiv(Et, 256), 256, 0, stream>>>(R.src, R.dst, R.E0, R.nSelf,
                                                  a_s, a_d, slope, w_e, m_buf);
      fix_m<<<cdiv(nm, 256), 256, 0, stream>>>(m_buf, nm);
      edge_expsum<<<cdiv(Et, 256), 256, 0, stream>>>(R.dst, R.E0, R.nSelf,
                                                     m_buf, w_e, den);
      float* acc = R.dstUser ? (sc == 0 ? acc_ul : acc_ug)
                             : (sc == 0 ? acc_il : acc_ig);
      const float* rg = (R.gateIdx >= 0) ? rel_gates : nullptr;
      edge_agg<<<cdiv((long long)Et * 32, 256), 256, 0, stream>>>(
          R.src, R.dst, R.E0, R.nSelf, w_e, den, h_s, rg, R.gateIdx, acc);
    }
  }

  // ---- fusion + output GEMMs ----
  // user (gate_W[0], out_W[0]); biases combined with softmax(rel_gates)
  build_cat<<<cdiv(nx, 256), 256, 0, stream>>>(acc_ul, acc_ug,
      bias_l + F, bias_l + 2 * F, bias_g + F, bias_g + 2 * F, rel_gates, cat_bf, N);
  gemm(cat_bf, gWt, gate_b, gpre, N, 2 * F);
  fuse_gate<<<cdiv(nx, 256), 256, 0, stream>>>(acc_ul, acc_ug,
      bias_l + F, bias_l + 2 * F, bias_g + F, bias_g + 2 * F, rel_gates, gpre, fused_bf, N);
  gemm(fused_bf, oWt, nullptr, (float*)d_out, N, F);

  // item (gate_W[1], out_W[1]); single relation -> unit gate
  build_cat<<<cdiv(nx, 256), 256, 0, stream>>>(acc_il, acc_ig,
      bias_l, nullptr, bias_g, nullptr, nullptr, cat_bf, N);
  gemm(cat_bf, gWt + (size_t)2 * F * F, gate_b + F, gpre, N, 2 * F);
  fuse_gate<<<cdiv(nx, 256), 256, 0, stream>>>(acc_il, acc_ig,
      bias_l, nullptr, bias_g, nullptr, nullptr, gpre, fused_bf, N);
  gemm(fused_bf, oWt + (size_t)F * F, nullptr, (float*)d_out + (size_t)N * F, N, F);
}
